// AttnPool_2d_67499706024356
// MI455X (gfx1250) — compile-verified, matrix operand order checked by reading, not run on hardware
//
#include <hip/hip_runtime.h>

// ---------------------------------------------------------------------------
// CDNA5 (gfx1250) implementation: windowed attention + MLP, all GEMMs on
// v_wmma_f32_16x16x32_bf16 (wave32). v2: every loaded B (weight) tile feeds
// two WMMAs (2 windows / 2 M-tiles per wave) to halve VMEM-per-WMMA.
// ---------------------------------------------------------------------------

typedef __attribute__((ext_vector_type(16))) __bf16 v16bf;
typedef __attribute__((ext_vector_type(8)))  __bf16 v8bf;
typedef __attribute__((ext_vector_type(8)))  float  v8f;

__device__ __forceinline__ unsigned short f2bfbits(float f) {
  unsigned u = __builtin_bit_cast(unsigned, f);
  u += 0x7FFFu + ((u >> 16) & 1u);           // round-to-nearest-even
  return (unsigned short)(u >> 16);
}
__device__ __forceinline__ __bf16 f2bf(float f) {
  unsigned short s = f2bfbits(f);
  return __builtin_bit_cast(__bf16, s);
}

// One WMMA 16-bit operand in the A-style layout (ISA 7.12.2): this lane
// supplies K = base..base+7 and base+16..base+23 (hi*8 already folded in).
__device__ __forceinline__ v16bf load_tile16(const unsigned short* p) {
  v8bf lo = *(const v8bf*)(p);
  v8bf hh = *(const v8bf*)(p + 16);
  return __builtin_shufflevector(lo, hh, 0, 1, 2, 3, 4, 5, 6, 7,
                                 8, 9, 10, 11, 12, 13, 14, 15);
}

__device__ __forceinline__ v8f wmma_bf(v16bf a, v16bf b, v8f c) {
  return __builtin_amdgcn_wmma_f32_16x16x32_bf16(false, a, false, b, (short)0,
                                                 c, false, false);
}

__device__ __forceinline__ float hsum8(v8f v) {
  return ((v[0] + v[1]) + (v[2] + v[3])) + ((v[4] + v[5]) + (v[6] + v[7]));
}

// ---------------------------------------------------------------------------
// fp32 -> bf16 weight conversion
// ---------------------------------------------------------------------------
__global__ void cvt_bf16_kernel(const float* __restrict__ src,
                                unsigned short* __restrict__ dst, int n) {
  int i = blockIdx.x * blockDim.x + threadIdx.x;
  if (i < n) dst[i] = f2bfbits(src[i]);
}

// ---------------------------------------------------------------------------
// Windowed attention: one wave = TWO 4x4 windows (two M=16 A-operands).
// Processed head-at-a-time: scores for head h need only head h's K columns,
// so score/attn state is 8 regs per window, reused across heads.
// ---------------------------------------------------------------------------
__global__ __launch_bounds__(256)
void attn_window_kernel(const float* __restrict__ x,
                        const unsigned short* __restrict__ WQb,
                        const unsigned short* __restrict__ WKb,
                        const unsigned short* __restrict__ WVb,
                        unsigned short* __restrict__ Zout) {
  const int wave = threadIdx.x >> 5;
  const int lane = threadIdx.x & 31;
  const int w0 = (blockIdx.x * 8 + wave) * 2;   // windows w0, w0+1
  const int m  = lane & 15;                     // A row / B (weight) row
  const int hi = lane >> 4;

  // ---- Preload A operands (both windows, bf16) for all 8 K-steps ----------
  v16bf A0[8], A1[8];
#pragma unroll
  for (int win = 0; win < 2; ++win) {
    const int w = w0 + win;
    const int bb = w >> 12;
    const int ij = w & 4095;
    const int wi = ij >> 6;
    const int wj = ij & 63;
    const float* xrow =
        x + (((size_t)bb * 256 + wi * 4 + (m >> 2)) * 256 + wj * 4 + (m & 3)) * 256;
#pragma unroll
    for (int k0 = 0; k0 < 8; ++k0) {
      const float* p = xrow + k0 * 32 + hi * 8;
      float4 f0 = *(const float4*)(p);
      float4 f1 = *(const float4*)(p + 4);
      float4 f2 = *(const float4*)(p + 16);
      float4 f3 = *(const float4*)(p + 20);
      v16bf a;
      a[0] = f2bf(f0.x);  a[1] = f2bf(f0.y);  a[2] = f2bf(f0.z);  a[3] = f2bf(f0.w);
      a[4] = f2bf(f1.x);  a[5] = f2bf(f1.y);  a[6] = f2bf(f1.z);  a[7] = f2bf(f1.w);
      a[8] = f2bf(f2.x);  a[9] = f2bf(f2.y);  a[10] = f2bf(f2.z); a[11] = f2bf(f2.w);
      a[12] = f2bf(f3.x); a[13] = f2bf(f3.y); a[14] = f2bf(f3.z); a[15] = f2bf(f3.w);
      if (win == 0) A0[k0] = a; else A1[k0] = a;
    }
  }

  unsigned short* zrow0 = Zout + (size_t)w0 * 512;
  unsigned short* zrow1 = zrow0 + 512;
  const size_t brow = (size_t)m * 256 + hi * 8;  // per-lane weight-row offset

#pragma unroll 1
  for (int h = 0; h < 8; ++h) {
    // ---- Q + K projections for head h, fused score accumulation ----------
    float p0[8] = {0.f, 0.f, 0.f, 0.f, 0.f, 0.f, 0.f, 0.f};
    float p1[8] = {0.f, 0.f, 0.f, 0.f, 0.f, 0.f, 0.f, 0.f};
#pragma unroll
    for (int t = 0; t < 4; ++t) {
      const size_t nbase = (size_t)(h * 4 + t) * 16 * 256 + brow;
      // q = columnMean(xw @ WQ^T): exact, mean commutes with the linear map
      v8f dq0 = {}, dq1 = {};
#pragma unroll
      for (int k0 = 0; k0 < 8; ++k0) {
        v16bf b = load_tile16(WQb + nbase + k0 * 32);
        dq0 = wmma_bf(A0[k0], b, dq0);
        dq1 = wmma_bf(A1[k0], b, dq1);
      }
      float qn0 = hsum8(dq0);
      float qn1 = hsum8(dq1);
      qn0 += __shfl_xor(qn0, 16, 32);  qn0 *= 0.0625f;
      qn1 += __shfl_xor(qn1, 16, 32);  qn1 *= 0.0625f;

      v8f dk0 = {}, dk1 = {};
#pragma unroll
      for (int k0 = 0; k0 < 8; ++k0) {
        v16bf b = load_tile16(WKb + nbase + k0 * 32);
        dk0 = wmma_bf(A0[k0], b, dk0);
        dk1 = wmma_bf(A1[k0], b, dk1);
      }
#pragma unroll
      for (int r = 0; r < 8; ++r) {
        p0[r] += qn0 * dk0[r];
        p1[r] += qn1 * dk1[r];
      }
    }
    // reduce the 64-wide head dot-product; lane<16 -> s=r, lane>=16 -> s=r+8
    float s0[8], s1[8];
#pragma unroll
    for (int r = 0; r < 8; ++r) {
      float v = p0[r];
      v += __shfl_xor(v, 1, 32); v += __shfl_xor(v, 2, 32);
      v += __shfl_xor(v, 4, 32); v += __shfl_xor(v, 8, 32);
      s0[r] = v * 0.125f;                       // 1/sqrt(QK=64)
      v = p1[r];
      v += __shfl_xor(v, 1, 32); v += __shfl_xor(v, 2, 32);
      v += __shfl_xor(v, 4, 32); v += __shfl_xor(v, 8, 32);
      s1[r] = v * 0.125f;
    }

    // ---- softmax over the 16 positions (both windows, in registers) -------
#pragma unroll
    for (int win = 0; win < 2; ++win) {
      float* s = win ? s1 : s0;
      float mx = s[0];
#pragma unroll
      for (int r = 1; r < 8; ++r) mx = fmaxf(mx, s[r]);
      mx = fmaxf(mx, __shfl_xor(mx, 16, 32));
      float sum = 0.f;
#pragma unroll
      for (int r = 0; r < 8; ++r) {
        float p = __expf(s[r] - mx);
        s[r] = p;
        sum += p;
      }
      sum += __shfl_xor(sum, 16, 32);
      float inv = 1.f / sum;
#pragma unroll
      for (int r = 0; r < 8; ++r) s[r] *= inv;
    }

    // ---- V projection for head h + attention-weighted combine -------------
#pragma unroll
    for (int t = 0; t < 4; ++t) {
      const int nt = h * 4 + t;
      const size_t nbase = (size_t)nt * 16 * 256 + brow;
      v8f dv0 = {}, dv1 = {};
#pragma unroll
      for (int k0 = 0; k0 < 8; ++k0) {
        v16bf b = load_tile16(WVb + nbase + k0 * 32);
        dv0 = wmma_bf(A0[k0], b, dv0);
        dv1 = wmma_bf(A1[k0], b, dv1);
      }
      float z0 = 0.f, z1 = 0.f;
#pragma unroll
      for (int r = 0; r < 8; ++r) {
        z0 += s0[r] * dv0[r];
        z1 += s1[r] * dv1[r];
      }
      z0 += __shfl_xor(z0, 16, 32);             // sum over both s-halves
      z1 += __shfl_xor(z1, 16, 32);
      if (lane < 16) {
        zrow0[nt * 16 + m] = f2bfbits(z0);
        zrow1[nt * 16 + m] = f2bfbits(z1);
      }
    }
  }
}

// ---------------------------------------------------------------------------
// Wave-tiled GEMM:  C[M,N] = act(A[M,K] @ W[N,K]^T + bias)   (bf16 in, WMMA)
// One wave owns a 32x32 output tile (2x2 WMMA tiles): per K-step, 4 A-loads
// + 4 B-loads feed 4 WMMAs (2 loads/WMMA). Bias+activation fused into store.
// ---------------------------------------------------------------------------
template <int RELU, int OUTF32>
__global__ __launch_bounds__(256)
void gemm_bias_kernel(const unsigned short* __restrict__ A,
                      const unsigned short* __restrict__ W,
                      const float* __restrict__ bias,
                      void* __restrict__ Cout, int M, int N, int K) {
  const int wave = threadIdx.x >> 5;
  const int lane = threadIdx.x & 31;
  const int task = blockIdx.x * (blockDim.x >> 5) + wave;
  const int nT = N >> 5;                       // 32-wide column chunks
  const int mt = task / nT;
  const int nt = task - mt * nT;
  if ((mt << 5) >= M) return;                  // wave-uniform; EXEC stays full
  const int m  = lane & 15;
  const int hi = lane >> 4;

  const unsigned short* a0 = A + (size_t)(mt * 32 + m) * K + hi * 8;
  const unsigned short* a1 = a0 + (size_t)16 * K;
  const unsigned short* w0 = W + (size_t)(nt * 32 + m) * K + hi * 8;
  const unsigned short* w1 = w0 + (size_t)16 * K;

  v8f acc[2][2] = {{{}, {}}, {{}, {}}};
  const int kSteps = K >> 5;
  for (int k0 = 0; k0 < kSteps; ++k0) {
    const int ko = k0 * 32;
    v16bf va0 = load_tile16(a0 + ko);
    v16bf va1 = load_tile16(a1 + ko);
    v16bf vb0 = load_tile16(w0 + ko);
    v16bf vb1 = load_tile16(w1 + ko);
    acc[0][0] = wmma_bf(va0, vb0, acc[0][0]);
    acc[0][1] = wmma_bf(va0, vb1, acc[0][1]);
    acc[1][0] = wmma_bf(va1, vb0, acc[1][0]);
    acc[1][1] = wmma_bf(va1, vb1, acc[1][1]);
  }

#pragma unroll
  for (int j = 0; j < 2; ++j) {
    const int cn = nt * 32 + j * 16 + m;       // D: lane holds column lane%16
    const float bv = bias[cn];
#pragma unroll
    for (int i = 0; i < 2; ++i) {
      const int rbase = mt * 32 + i * 16 + hi * 8;  // D: VGPR r -> row r(+8)
#pragma unroll
      for (int r = 0; r < 8; ++r) {
        float v = acc[i][j][r] + bv;
        if (RELU) v = fmaxf(v, 0.f);
        if (OUTF32)
          ((float*)Cout)[(size_t)(rbase + r) * N + cn] = v;
        else
          ((unsigned short*)Cout)[(size_t)(rbase + r) * N + cn] = f2bfbits(v);
      }
    }
  }
}

// ---------------------------------------------------------------------------
extern "C" void kernel_launch(void* const* d_in, const int* in_sizes, int n_in,
                              void* d_out, int out_size, void* d_ws,
                              size_t ws_size, hipStream_t stream) {
  (void)in_sizes; (void)n_in; (void)out_size; (void)ws_size;

  const float* x   = (const float*)d_in[0];
  const float* WQ  = (const float*)d_in[1];
  const float* WK  = (const float*)d_in[2];
  const float* WV  = (const float*)d_in[3];
  const float* WZw = (const float*)d_in[4];
  const float* WZb = (const float*)d_in[5];
  const float* Ew  = (const float*)d_in[6];
  const float* Ebi = (const float*)d_in[7];
  const float* Rw  = (const float*)d_in[8];
  const float* Rbi = (const float*)d_in[9];

  // Workspace layout (bf16 elements).
  unsigned short* ws   = (unsigned short*)d_ws;
  unsigned short* WQ16 = ws;                     // 512*256
  unsigned short* WK16 = WQ16 + 512 * 256;       // 512*256
  unsigned short* WV16 = WK16 + 512 * 256;       // 512*256
  unsigned short* WZ16 = WV16 + 512 * 256;       // 256*512
  unsigned short* E16  = WZ16 + 256 * 512;       // 1024*256
  unsigned short* R16  = E16  + 1024 * 256;      // 256*1024
  unsigned short* Zb   = R16  + 256 * 1024;      // 16384*512
  unsigned short* Tb   = Zb   + 16384 * 512;     // 16384*256
  unsigned short* Hb   = Tb   + 16384 * 256;     // 16384*1024

  // 1) weights -> bf16 (L2-resident, ~2 MB total)
  auto cvt = [&](const float* s, unsigned short* d, int n) {
    cvt_bf16_kernel<<<(n + 255) / 256, 256, 0, stream>>>(s, d, n);
  };
  cvt(WQ,  WQ16, 512 * 256);
  cvt(WK,  WK16, 512 * 256);
  cvt(WV,  WV16, 512 * 256);
  cvt(WZw, WZ16, 256 * 512);
  cvt(Ew,  E16,  1024 * 256);
  cvt(Rw,  R16,  256 * 1024);

  // 2) windowed attention: 16384 windows, 8 waves x 2 windows per block
  attn_window_kernel<<<1024, 256, 0, stream>>>(x, WQ16, WK16, WV16, Zb);

  // 3) MLP head: M = 16384 rows (one per window); 32x32 tile per wave
  // z @ WZ^T + b        : [16384,512]x[256,512]^T -> [16384,256]; 4096 tiles
  gemm_bias_kernel<0, 0><<<(512 * 8) / 8, 256, 0, stream>>>(
      Zb, WZ16, WZb, (void*)Tb, 16384, 256, 512);
  // relu(t @ E^T + b)   : -> [16384,1024]; 16384 tiles
  gemm_bias_kernel<1, 0><<<(512 * 32) / 8, 256, 0, stream>>>(
      Tb, E16, Ebi, (void*)Hb, 16384, 1024, 256);
  // relu(h @ R^T + b)   : -> [16384,256] fp32 final output; 4096 tiles
  gemm_bias_kernel<1, 1><<<(512 * 8) / 8, 256, 0, stream>>>(
      Hb, R16, Rbi, d_out, 16384, 256, 1024);
}
